// TemporalGCN_86526411145513
// MI455X (gfx1250) — matrix operand order repaired and checked by reading, not compiled, not run on hardware
//
#include <hip/hip_runtime.h>
#include <hip/hip_bf16.h>

// ---------------------------------------------------------------------------
// TemporalGCN for MI455X (gfx1250, wave32, WMMA).
// Pipeline:
//   conv1+relu+pool (f32)  -> h1  [B,16,1024]
//   conv2+relu+pool (f32)  -> X0  [N,32]  f16 row-major (WMMA-A friendly)
//   GCN1: WMMA f16 GEMM [N,32]x[32,128] -> Ya f16; chain-smooth+bias+relu -> Xb
//   GCN2: WMMA f16 GEMM [N,128]x[128,128] -> Ya;  chain-smooth+bias+relu -> Xb
//   mean over time + FC -> out [B,64] f32
// ---------------------------------------------------------------------------

typedef __attribute__((ext_vector_type(16))) _Float16 v16h;
typedef __attribute__((ext_vector_type(8)))  _Float16 v8h;
typedef __attribute__((ext_vector_type(4)))  _Float16 v4h;
typedef __attribute__((ext_vector_type(8)))  float    v8f;

#define B_    1024
#define CIN_  9
#define T_    2048
#define RT_   512
#define NN_   (B_ * RT_)   /* 524288 nodes */
#define HID_  128
#define OUT_  64

union Frag16 {            // one WMMA 16-bit operand: 8 VGPRs per lane
  v16h v;
  v8h  h[2];
};

// ---------------------------------------------------------------------------
// Kernel 1: conv1d(9->16, k=5, pad=2) + ReLU + MaxPool2  -> h1 [B,16,1024] f32
// One thread per output element. relu(max(a,b)) == max(relu a, relu b),
// and initializing the max with 0 applies the ReLU for free.
// ---------------------------------------------------------------------------
__global__ __launch_bounds__(256) void conv1_pool(
    const float* __restrict__ x, const float* __restrict__ w,
    const float* __restrict__ bias, float* __restrict__ h1) {
  int idx = blockIdx.x * blockDim.x + threadIdx.x;
  if (idx >= B_ * 16 * 1024) return;
  int t  = idx & 1023;
  int co = (idx >> 10) & 15;
  int bb = idx >> 14;
  const float* xb = x + (size_t)bb * CIN_ * T_;
  float m = 0.f;
#pragma unroll
  for (int p = 0; p < 2; ++p) {
    int tc = 2 * t + p;
    float s = bias[co];
    for (int ci = 0; ci < CIN_; ++ci) {
      const float* xr = xb + (size_t)ci * T_;
      const float* wr = w + (co * CIN_ + ci) * 5;
#pragma unroll
      for (int k = 0; k < 5; ++k) {
        int u = tc + k - 2;
        if (u >= 0 && u < T_) s += xr[u] * wr[k];
      }
    }
    m = fmaxf(m, s);
  }
  h1[idx] = m;
}

// ---------------------------------------------------------------------------
// Kernel 2: conv1d(16->32, k=5, pad=2) + ReLU + MaxPool2, transposed into
// node-feature layout X0[(b*512+t)*32 + c] as f16 (row-major [N,32]).
// ---------------------------------------------------------------------------
__global__ __launch_bounds__(256) void conv2_pool(
    const float* __restrict__ h1, const float* __restrict__ w,
    const float* __restrict__ bias, _Float16* __restrict__ X0) {
  int idx = blockIdx.x * blockDim.x + threadIdx.x;
  if (idx >= B_ * 32 * 512) return;
  int t  = idx & 511;
  int co = (idx >> 9) & 31;
  int bb = idx >> 14;
  const float* hb = h1 + (size_t)bb * 16 * 1024;
  float m = 0.f;
#pragma unroll
  for (int p = 0; p < 2; ++p) {
    int tc = 2 * t + p;
    float s = bias[co];
    for (int ci = 0; ci < 16; ++ci) {
      const float* xr = hb + (size_t)ci * 1024;
      const float* wr = w + (co * 16 + ci) * 5;
#pragma unroll
      for (int k = 0; k < 5; ++k) {
        int u = tc + k - 2;
        if (u >= 0 && u < 1024) s += xr[u] * wr[k];
      }
    }
    m = fmaxf(m, s);
  }
  X0[((size_t)bb * RT_ + t) * 32 + co] = (_Float16)m;
}

// ---------------------------------------------------------------------------
// Kernel 3: GEMM  Y[N,128] = X[N,KIN] @ W[KIN,128]  via v_wmma_f32_16x16x32_f16.
// - A fragment (16x32, ISA layout): lane L holds row M=L&15, K in
//   {koff..koff+7} U {16+koff..16+koff+7}, koff = (L>=16)?8:0  ->
//   two contiguous b128 loads from the row-major f16 feature matrix.
// - B fragments are pre-converted to f16 in LDS in fragment order
//   (B 32x16: lane L holds column N=L&15, same K order as A fragment),
//   so each lane reads its operand with two ds_load_b128.
// - Each wave computes one 16-row stripe across all 8 column tiles.
//   Grid covers N/16 row tiles exactly: EXEC all-1s (WMMA requirement).
// ---------------------------------------------------------------------------
template <int KIN>
__global__ __launch_bounds__(256) void gcn_gemm(
    const _Float16* __restrict__ X, const float* __restrict__ W,
    _Float16* __restrict__ Y) {
  constexpr int KT = KIN / 32;
  __shared__ _Float16 ldsB[KT * 8 * 32 * 16];

  // Stage weight fragments (f32 -> f16) in LDS, fragment-major.
  for (int idx = threadIdx.x; idx < KT * 8 * 32 * 16; idx += 256) {
    int j    = idx & 15;
    int lane = (idx >> 4) & 31;
    int frag = idx >> 9;
    int kt = frag >> 3, nt = frag & 7;
    int k = kt * 32 + (j & 7) + ((j & 8) ? 16 : 0) + ((lane & 16) ? 8 : 0);
    int n = nt * 16 + (lane & 15);
    ldsB[idx] = (_Float16)W[k * HID_ + n];
  }
  __syncthreads();

  const int lane = threadIdx.x & 31;
  const int tile = blockIdx.x * 8 + (threadIdx.x >> 5);  // 16-row tile index
  const int row  = tile * 16 + (lane & 15);
  const int koff = (lane & 16) ? 8 : 0;

  Frag16 a[KT];
#pragma unroll
  for (int kt = 0; kt < KT; ++kt) {
    const _Float16* p = X + (size_t)row * KIN + kt * 32 + koff;
    a[kt].h[0] = *(const v8h*)p;          // global_load_b128
    a[kt].h[1] = *(const v8h*)(p + 16);   // global_load_b128
  }

#pragma unroll
  for (int nt = 0; nt < 8; ++nt) {
    v8f acc = {};
#pragma unroll
    for (int kt = 0; kt < KT; ++kt) {
      Frag16 bF;
      const _Float16* lp = &ldsB[(size_t)((kt * 8 + nt) * 32 + lane) * 16];
      bF.h[0] = *(const v8h*)lp;          // ds_load_b128
      bF.h[1] = *(const v8h*)(lp + 8);    // ds_load_b128
      acc = __builtin_amdgcn_wmma_f32_16x16x32_f16(
          /*neg_a=*/false, a[kt].v, /*neg_b=*/false, bF.v,
          /*c_mod=*/(short)0, acc, /*reuse_a=*/false, /*reuse_b=*/false);
    }
    // D layout: lane L VGPR r -> (M = r + (L>=16?8:0), N = L&15)
    const int n0 = nt * 16 + (lane & 15);
    const int r0 = (lane & 16) ? 8 : 0;
#pragma unroll
    for (int r = 0; r < 8; ++r)
      Y[(size_t)(tile * 16 + r0 + r) * HID_ + n0] = (_Float16)acc[r];
  }
}

// ---------------------------------------------------------------------------
// Kernel 4: GCN chain aggregation + bias + ReLU.
// out[i] = relu( dinv_i^2 * h[i] + dinv_i*dinv_{i-1}*h[i-1]
//              + dinv_i*dinv_{i+1}*h[i+1] + b )   within each sample's chain.
// deg = 3 interior, 2 at the chain endpoints (self-loops included).
// Each thread handles 4 features (v4h = 8B loads).
// ---------------------------------------------------------------------------
__global__ __launch_bounds__(256) void gcn_smooth(
    const _Float16* __restrict__ Y, const float* __restrict__ bias,
    _Float16* __restrict__ Xo) {
  int idx = blockIdx.x * blockDim.x + threadIdx.x;
  if (idx >= NN_ * 32) return;
  int f0 = (idx & 31) * 4;
  int i  = idx >> 5;
  int t  = i & (RT_ - 1);
  const float rs2 = 0.70710678118654752f;   // 1/sqrt(2)
  const float rs3 = 0.57735026918962576f;   // 1/sqrt(3)
  float di = (t == 0 || t == RT_ - 1) ? rs2 : rs3;

  v4h c = *(const v4h*)(Y + (size_t)i * HID_ + f0);
  float acc[4];
#pragma unroll
  for (int j = 0; j < 4; ++j) acc[j] = (float)c[j] * di * di + bias[f0 + j];

  if (t > 0) {
    float dn = di * ((t - 1 == 0 || t - 1 == RT_ - 1) ? rs2 : rs3);
    v4h p = *(const v4h*)(Y + (size_t)(i - 1) * HID_ + f0);
#pragma unroll
    for (int j = 0; j < 4; ++j) acc[j] += (float)p[j] * dn;
  }
  if (t < RT_ - 1) {
    float dn = di * ((t + 1 == 0 || t + 1 == RT_ - 1) ? rs2 : rs3);
    v4h p = *(const v4h*)(Y + (size_t)(i + 1) * HID_ + f0);
#pragma unroll
    for (int j = 0; j < 4; ++j) acc[j] += (float)p[j] * dn;
  }
  v4h o;
#pragma unroll
  for (int j = 0; j < 4; ++j) o[j] = (_Float16)fmaxf(acc[j], 0.f);
  *(v4h*)(Xo + (size_t)i * HID_ + f0) = o;
}

// ---------------------------------------------------------------------------
// Kernel 5: temporal mean (over 512 nodes/sample) + FC 128->64.
// One block (128 threads) per sample.
// ---------------------------------------------------------------------------
__global__ __launch_bounds__(128) void mean_fc(
    const _Float16* __restrict__ X, const float* __restrict__ fw,
    const float* __restrict__ fb, float* __restrict__ out) {
  __shared__ float m[HID_];
  int bb = blockIdx.x;
  int f  = threadIdx.x;
  const _Float16* p = X + (size_t)bb * RT_ * HID_ + f;
  float s = 0.f;
  for (int t = 0; t < RT_; ++t) s += (float)p[(size_t)t * HID_];
  m[f] = s * (1.0f / RT_);
  __syncthreads();
  if (f < OUT_) {
    float o = fb[f];
#pragma unroll 4
    for (int k = 0; k < HID_; ++k) o += m[k] * fw[k * OUT_ + f];
    out[(size_t)bb * OUT_ + f] = o;
  }
}

// ---------------------------------------------------------------------------
// Host launcher. Workspace layout (352 MB total):
//   [0,   64MB)  h1  f32 [B,16,1024]
//   [64,  96MB)  X0  f16 [N,32]
//   [96, 224MB)  Ya  f16 [N,128]  (GEMM outputs, ping)
//   [224,352MB)  Xb  f16 [N,128]  (smoothed activations, pong)
// ---------------------------------------------------------------------------
extern "C" void kernel_launch(void* const* d_in, const int* in_sizes, int n_in,
                              void* d_out, int out_size, void* d_ws, size_t ws_size,
                              hipStream_t stream) {
  (void)in_sizes; (void)n_in; (void)out_size; (void)ws_size;
  const float* x   = (const float*)d_in[0];
  const float* c1w = (const float*)d_in[1];
  const float* c1b = (const float*)d_in[2];
  const float* c2w = (const float*)d_in[3];
  const float* c2b = (const float*)d_in[4];
  const float* g1w = (const float*)d_in[5];
  const float* g1b = (const float*)d_in[6];
  const float* g2w = (const float*)d_in[7];
  const float* g2b = (const float*)d_in[8];
  const float* fcw = (const float*)d_in[9];
  const float* fcb = (const float*)d_in[10];
  float* out = (float*)d_out;

  char* ws = (char*)d_ws;
  float*    h1 = (float*)ws;
  _Float16* X0 = (_Float16*)(ws + ((size_t)64  << 20));
  _Float16* Ya = (_Float16*)(ws + ((size_t)96  << 20));
  _Float16* Xb = (_Float16*)(ws + ((size_t)224 << 20));

  conv1_pool<<<(B_ * 16 * 1024) / 256, 256, 0, stream>>>(x, c1w, c1b, h1);
  conv2_pool<<<(B_ * 32 * 512) / 256, 256, 0, stream>>>(h1, c2w, c2b, X0);

  // GCN layer 1: [N,32] @ [32,128]
  gcn_gemm<32><<<NN_ / 16 / 8, 256, 0, stream>>>(X0, g1w, Ya);
  gcn_smooth<<<(NN_ * 32) / 256, 256, 0, stream>>>(Ya, g1b, Xb);

  // GCN layer 2: [N,128] @ [128,128]
  gcn_gemm<128><<<NN_ / 16 / 8, 256, 0, stream>>>(Xb, g2w, Ya);
  gcn_smooth<<<(NN_ * 32) / 256, 256, 0, stream>>>(Ya, g2b, Xb);

  mean_fc<<<B_, 128, 0, stream>>>(Xb, fcw, fcb, out);
}
